// VAE_mu_var_25864293057260
// MI455X (gfx1250) — matrix operand order repaired and checked by reading, not compiled, not run on hardware
//
#include <hip/hip_runtime.h>
#include <hip/hip_bf16.h>
#include <math.h>

typedef __attribute__((ext_vector_type(16))) __bf16 v16bf;
typedef __attribute__((ext_vector_type(8)))  float  v8f;

#define B_SZ 65536
#define D_SZ 784
#define H_SZ 392
#define L_SZ 32
#define HP   416     // H padded (mult of 32)
#define DP   800     // D padded as K of gemm1 (mult of 32)
#define XSTRIDE 808  // padded LDS strides: (stride/2) % 64 == 20 -> conflict-free b128
#define HSTRIDE 424
#define ZSTRIDE 40
#define MT 64        // batch rows per block

// ---------- helpers ----------
__device__ __forceinline__ unsigned int f2bf(float f) {
  unsigned int u = __builtin_bit_cast(unsigned int, f);
  u += 0x7FFFu + ((u >> 16) & 1u);   // round-to-nearest-even
  return u >> 16;
}

// A fragment (16x32 bf16) from LDS, ISA layout:
// lanes 0-15: M=lane, K in {0..7, 16..23}; lanes 16-31: M=lane-16, K+8.
__device__ __forceinline__ v16bf lds_afrag(const unsigned short* p0, int stride, int lane) {
  int row = lane & 15;
  int kh  = (lane >> 4) << 3;       // 0 or 8
  const unsigned short* p = p0 + row * stride + kh;
  union { uint4 q[2]; v16bf v; } u;
  u.q[0] = *(const uint4*)(p);        // K [kh, kh+8)
  u.q[1] = *(const uint4*)(p + 16);   // K [16+kh, 16+kh+8)
  return u.v;
}

// B fragment (32x16 bf16) pre-packed in global: dword (frag*256 + lane*8 + j)
__device__ __forceinline__ v16bf g_bfrag(const unsigned int* pw, int frag, int lane) {
  const unsigned int* p = pw + ((size_t)frag << 8) + (lane << 3);
  union { uint4 q[2]; v16bf v; } u;
  u.q[0] = *(const uint4*)(p);
  u.q[1] = *(const uint4*)(p + 4);
  return u.v;
}

#define WMMA_BF16(a,b,c) __builtin_amdgcn_wmma_f32_16x16x32_bf16(false,(a),false,(b),(short)0,(c),false,false)

// ---------- weight packer: W (N x K, f32 row-major) -> B-fragment bf16 layout ----------
// frag f = nt*(Kp/32)+kt ; dword(lane,j) = pack(W[n][k], W[n][k+1]),
// n = nt*16 + (lane&15), k = kt*32 + ((lane>=16)?8:0) + ((j>=4)?8:0) + 2j   (zero pad OOB)
__global__ void pack_w(const float* __restrict__ W, unsigned int* __restrict__ dst,
                       int N, int K, int Np, int Kp) {
  int total = (Np * Kp) >> 1;
  int nKt = Kp >> 5;
  for (int idx = blockIdx.x * blockDim.x + threadIdx.x; idx < total;
       idx += gridDim.x * blockDim.x) {
    int f = idx >> 8, r = idx & 255;
    int lane = r >> 3, j = r & 7;
    int nt = f / nKt, kt = f - nt * nKt;
    int n = nt * 16 + (lane & 15);
    int k = kt * 32 + ((lane >> 4) << 3) + ((j >> 2) << 3) + (j << 1);
    float f0 = (n < N && k     < K) ? W[(size_t)n * K + k    ] : 0.0f;
    float f1 = (n < N && (k+1) < K) ? W[(size_t)n * K + k + 1] : 0.0f;
    dst[idx] = f2bf(f0) | (f2bf(f1) << 16);
  }
}

__global__ void zero_acc(float* acc) { if (threadIdx.x < 2) acc[threadIdx.x] = 0.0f; }

__global__ void finalize(const float* __restrict__ acc, float* __restrict__ outs) {
  // nll = 0.5*D*log(2pi) + nll_sum/B ; scalar = nll + 0.5*kl_sum
  outs[0] = 0.5f * 784.0f * 1.8378770664093453f
          + acc[1] * (1.0f / 65536.0f)
          + 0.5f * acc[0];
}

// ---------- fused VAE forward ----------
__global__ __launch_bounds__(256)
void vae_fused(const float* __restrict__ x, const float* __restrict__ eps_z,
               const float* __restrict__ eps_y,
               const float* __restrict__ b_e1, const float* __restrict__ b_emu,
               const float* __restrict__ b_elv, const float* __restrict__ b_d1,
               const float* __restrict__ b_dmu, const float* __restrict__ b_dlv,
               const unsigned int* __restrict__ pW_e1, const unsigned int* __restrict__ pW_emu,
               const unsigned int* __restrict__ pW_elv, const unsigned int* __restrict__ pW_d1,
               const unsigned int* __restrict__ pW_dmu, const unsigned int* __restrict__ pW_dlv,
               float* __restrict__ out, float* __restrict__ acc) {
  __shared__ unsigned short x_lds[MT * XSTRIDE];  // 103424 B
  __shared__ unsigned short h_lds[MT * HSTRIDE];  //  54272 B (also hd)
  __shared__ unsigned short z_lds[MT * ZSTRIDE];  //   5120 B

  const int tid  = threadIdx.x;
  const int w    = tid >> 5;
  const int lane = tid & 31;
  const int ws_row = (w & 3) << 4;   // 16-row window of this wave
  const int par    = w >> 2;         // n-tile parity (0/1)
  const size_t m0  = (size_t)blockIdx.x * MT;

  float kl_acc = 0.0f, nll_acc = 0.0f;

  // ---- stage x tile -> bf16 LDS (K padded to 800 with zeros) ----
  for (int i = tid; i < MT * 12; i += 256) {             // pad cols 784..807
    int r = i / 12, c = i - r * 12;
    *(unsigned int*)&x_lds[r * XSTRIDE + 784 + c * 2] = 0u;
  }
  for (int i = tid; i < MT * 196; i += 256) {            // 784 = 196 float4 / row
    int r = i / 196, c4 = i - r * 196;
    float4 f = *(const float4*)(x + (m0 + r) * D_SZ + c4 * 4);
    uint2 u;
    u.x = f2bf(f.x) | (f2bf(f.y) << 16);
    u.y = f2bf(f.z) | (f2bf(f.w) << 16);
    *(uint2*)&x_lds[r * XSTRIDE + c4 * 4] = u;
  }
  __syncthreads();

  const int mh = (lane >> 4) << 3;   // C/D fragment row-half offset
  const int nlane = lane & 15;

  // ---- GEMM1: h = relu(x @ W_e1^T + b_e1)  [N=416 padded, K=800 padded] ----
  for (int nt = par; nt < 26; nt += 2) {
    v8f accv = {};
    const unsigned short* xb = x_lds + ws_row * XSTRIDE;
    for (int kt = 0; kt < 25; ++kt) {
      v16bf a = lds_afrag(xb + kt * 32, XSTRIDE, lane);
      v16bf b = g_bfrag(pW_e1, nt * 25 + kt, lane);
      accv = WMMA_BF16(a, b, accv);
    }
    int n = nt * 16 + nlane;
    float bias = (n < H_SZ) ? b_e1[n] : 0.0f;
#pragma unroll
    for (int r = 0; r < 8; ++r) {
      float v = fmaxf(accv[r] + bias, 0.0f);
      h_lds[(ws_row + mh + r) * HSTRIDE + n] = (unsigned short)f2bf(v);
    }
  }
  __syncthreads();

  // ---- GEMM2/3: mu_z, logvar_z ; z = mu + eps*exp(0.5*lv) ; KL terms ----
  {
    const int nt = par;  // N = 32 -> each parity does one 16-col tile
    v8f amu = {}, alv = {};
    const unsigned short* hb = h_lds + ws_row * HSTRIDE;
    for (int kt = 0; kt < 13; ++kt) {
      v16bf a  = lds_afrag(hb + kt * 32, HSTRIDE, lane);
      v16bf bm = g_bfrag(pW_emu, nt * 13 + kt, lane);
      v16bf bl = g_bfrag(pW_elv, nt * 13 + kt, lane);
      amu = WMMA_BF16(a, bm, amu);
      alv = WMMA_BF16(a, bl, alv);
    }
    int n = nt * 16 + nlane;
    float bm = b_emu[n], bl = b_elv[n];
#pragma unroll
    for (int r = 0; r < 8; ++r) {
      float mu = fmaxf(amu[r] + bm, 0.0f);
      float lv = fmaxf(alv[r] + bl, 0.0f);
      float ev = __expf(lv);
      kl_acc += mu * mu + ev - __logf(1e-8f + ev) - 1.0f;
      int row = ws_row + mh + r;
      float ez = eps_z[(m0 + row) * L_SZ + n];
      float zv = mu + ez * __expf(0.5f * lv);
      z_lds[row * ZSTRIDE + n] = (unsigned short)f2bf(zv);
    }
  }
  __syncthreads();

  // ---- GEMM4: hd = relu(z @ W_d1^T + b_d1)  [K=32, one k-tile] ----
  {
    v16bf az = lds_afrag(z_lds + ws_row * ZSTRIDE, ZSTRIDE, lane);
    for (int nt = par; nt < 26; nt += 2) {
      v8f accv = {};
      v16bf b = g_bfrag(pW_d1, nt, lane);
      accv = WMMA_BF16(az, b, accv);
      int n = nt * 16 + nlane;
      float bias = (n < H_SZ) ? b_d1[n] : 0.0f;
#pragma unroll
      for (int r = 0; r < 8; ++r) {
        float v = fmaxf(accv[r] + bias, 0.0f);
        h_lds[(ws_row + mh + r) * HSTRIDE + n] = (unsigned short)f2bf(v);
      }
    }
  }
  __syncthreads();

  // ---- GEMM5/6: mu_y, logvar_y ; image = sigmoid(mu+eps*sd) ; NLL terms ----
  for (int nt = par; nt < 49; nt += 2) {
    v8f amu = {}, alv = {};
    const unsigned short* hb = h_lds + ws_row * HSTRIDE;
    for (int kt = 0; kt < 13; ++kt) {
      v16bf a  = lds_afrag(hb + kt * 32, HSTRIDE, lane);
      v16bf bm = g_bfrag(pW_dmu, nt * 13 + kt, lane);
      v16bf bl = g_bfrag(pW_dlv, nt * 13 + kt, lane);
      amu = WMMA_BF16(a, bm, amu);
      alv = WMMA_BF16(a, bl, alv);
    }
    int n = nt * 16 + nlane;
    float bm = b_dmu[n], bl = b_dlv[n];
#pragma unroll
    for (int r = 0; r < 8; ++r) {
      size_t row = m0 + ws_row + mh + r;
      size_t idx = row * D_SZ + n;
      float mu = fmaxf(amu[r] + bm, 0.0f);
      float lv = fmaxf(alv[r] + bl, 0.0f);
      float xv = x[idx];
      float ey = eps_y[idx];
      float sd = __expf(0.5f * lv);          // reference: var = exp(0.5*logvar)
      float t  = mu + ey * sd;
      out[idx] = 1.0f / (1.0f + __expf(-t)); // sigmoid
      float d  = xv - mu;
      nll_acc += 0.5f * (0.5f * lv + d * d / sd);
    }
  }

  // ---- wave reduce + atomics ----
#pragma unroll
  for (int off = 16; off > 0; off >>= 1) {
    kl_acc  += __shfl_xor(kl_acc,  off, 32);
    nll_acc += __shfl_xor(nll_acc, off, 32);
  }
  if (lane == 0) {
    atomicAdd(&acc[0], kl_acc);
    atomicAdd(&acc[1], nll_acc);
  }
}

// ---------- host ----------
extern "C" void kernel_launch(void* const* d_in, const int* in_sizes, int n_in,
                              void* d_out, int out_size, void* d_ws, size_t ws_size,
                              hipStream_t stream) {
  const float* x     = (const float*)d_in[0];
  const float* eps_z = (const float*)d_in[1];
  const float* eps_y = (const float*)d_in[2];
  const float* W_e1  = (const float*)d_in[3];
  const float* b_e1  = (const float*)d_in[4];
  const float* W_emu = (const float*)d_in[5];
  const float* b_emu = (const float*)d_in[6];
  const float* W_elv = (const float*)d_in[7];
  const float* b_elv = (const float*)d_in[8];
  const float* W_d1  = (const float*)d_in[9];
  const float* b_d1  = (const float*)d_in[10];
  const float* W_dmu = (const float*)d_in[11];
  const float* b_dmu = (const float*)d_in[12];
  const float* W_dlv = (const float*)d_in[13];
  const float* b_dlv = (const float*)d_in[14];

  unsigned int* wsu = (unsigned int*)d_ws;
  float* accp = (float*)d_ws;                    // [0]=kl_sum [1]=nll_sum
  unsigned int* pe1  = wsu + 16;                 // 416*800/2 = 166400 dwords
  unsigned int* pemu = pe1  + 166400;            // 32*416/2  =   6656
  unsigned int* pelv = pemu + 6656;
  unsigned int* pd1  = pelv + 6656;              // 416*32/2  =   6656
  unsigned int* pdmu = pd1  + 6656;              // 784*416/2 = 163072
  unsigned int* pdlv = pdmu + 163072;

  zero_acc<<<1, 32, 0, stream>>>(accp);
  pack_w<<<256, 256, 0, stream>>>(W_e1,  pe1,  H_SZ, D_SZ, HP, DP);
  pack_w<<<64,  256, 0, stream>>>(W_emu, pemu, L_SZ, H_SZ, L_SZ, HP);
  pack_w<<<64,  256, 0, stream>>>(W_elv, pelv, L_SZ, H_SZ, L_SZ, HP);
  pack_w<<<64,  256, 0, stream>>>(W_d1,  pd1,  H_SZ, L_SZ, HP, L_SZ);
  pack_w<<<256, 256, 0, stream>>>(W_dmu, pdmu, D_SZ, H_SZ, D_SZ, HP);
  pack_w<<<256, 256, 0, stream>>>(W_dlv, pdlv, D_SZ, H_SZ, D_SZ, HP);

  vae_fused<<<B_SZ / MT, 256, 0, stream>>>(
      x, eps_z, eps_y, b_e1, b_emu, b_elv, b_d1, b_dmu, b_dlv,
      pe1, pemu, pelv, pd1, pdmu, pdlv, (float*)d_out, accp);

  finalize<<<1, 1, 0, stream>>>(accp, (float*)d_out + (size_t)B_SZ * D_SZ);
}
